// Attention_59296318489273
// MI455X (gfx1250) — matrix operand order repaired and checked by reading, not compiled, not run on hardware
//
#include <hip/hip_runtime.h>
#include <hip/hip_bf16.h>

// ---------------------------------------------------------------------------
// Causal multi-head attention for MI455X (gfx1250, wave32, WMMA bf16->f32)
//   B=2, S=2048, D_MODEL=1024, H=16, D_HEAD=64
// Pipeline:
//   cvt (x -> bf16, W -> fragment-native bf16 layout)
//   -> QKV GEMM (async-LDS double-buffered W tiles, writes Q,K [s][e], V^T [e][s])
//   -> register-resident causal flash attention (WMMA, shfl-only softmax)
//   -> output projection GEMM (async-LDS double-buffered W_O tiles, f32 out)
// ---------------------------------------------------------------------------

typedef __attribute__((ext_vector_type(16))) __bf16 v16bf;
typedef __attribute__((ext_vector_type(8)))  __bf16 v8bf;
typedef __attribute__((ext_vector_type(8)))  float  v8f;

#define Bq   2
#define Sq   2048
#define Dq   1024
#define Hq   16
#define Eq   64
#define BSq  (Bq * Sq)          // 4096 rows
#define IGNORE_VAL (-100000.0f)

__device__ __forceinline__ v8f wmma_bf16(v16bf a, v16bf b, v8f c) {
  return __builtin_amdgcn_wmma_f32_16x16x32_bf16(
      /*neg_a=*/false, a, /*neg_b=*/false, b,
      /*c_mod=*/(short)0, c, /*reuse_a=*/false, /*reuse_b=*/false);
}

// A fragment (16x32 bf16, M x K), source row-major [row][k], leading dim ld.
// Layout: row = lane&15; lane<16 holds K {0..15}, lane>=16 {16..31}
__device__ __forceinline__ v16bf load_a(const __bf16* __restrict__ p, int ld,
                                        int row0, int k0, int lane) {
  int m = lane & 15, kh = lane >> 4;
  const __bf16* r = p + (size_t)(row0 + m) * ld + k0 + kh * 16;
  v8bf lo = *(const v8bf*)(r);
  v8bf hi = *(const v8bf*)(r + 8);
  return __builtin_shufflevector(lo, hi, 0, 1, 2, 3, 4, 5, 6, 7,
                                 8, 9, 10, 11, 12, 13, 14, 15);
}

// B fragment where B = M^T of an N-major source: element(k,n) = p[(col0+n)*ld + k0+k]
// (used for Q^T: one contiguous 32-byte load per lane)
__device__ __forceinline__ v16bf load_bT(const __bf16* __restrict__ p, int ld,
                                         int k0, int col0, int lane) {
  int n = lane & 15, kh = lane >> 4;
  return *(const v16bf*)(p + (size_t)(col0 + n) * ld + k0 + kh * 16);
}

// Issue one async global->LDS 16B copy for this lane (CDNA5 async path).
__device__ __forceinline__ void async_copy16(uint32_t lds_byte_addr,
                                             const void* gaddr) {
  asm volatile("global_load_async_to_lds_b128 %0, %1, off"
               :: "v"(lds_byte_addr), "v"((uint64_t)(uintptr_t)gaddr)
               : "memory");
}
__device__ __forceinline__ void wait_async0() {
  asm volatile("s_wait_asynccnt 0x0" ::: "memory");
}

// ---------------------------------------------------------------------------
// Kernel 0a: f32 -> bf16 linear conversion (x)
// ---------------------------------------------------------------------------
__global__ void cvt_bf16_kernel(const float4* __restrict__ in,
                                __bf16* __restrict__ out, int n4) {
  int i = blockIdx.x * blockDim.x + threadIdx.x;
  if (i < n4) {
    float4 v = in[i];
    out[4 * i + 0] = (__bf16)v.x;
    out[4 * i + 1] = (__bf16)v.y;
    out[4 * i + 2] = (__bf16)v.z;
    out[4 * i + 3] = (__bf16)v.w;
  }
}

// ---------------------------------------------------------------------------
// Kernel 0b: f32 weights [head][K][N] -> fragment-native bf16 [head][K/16][N][16]
//   so a WMMA B-fragment is 16 contiguous bf16 per lane.
// ---------------------------------------------------------------------------
__global__ void cvt_wfrag_kernel(const float* __restrict__ src,
                                 __bf16* __restrict__ dst, int K, int N) {
  int idx = blockIdx.x * blockDim.x + threadIdx.x;
  int head = blockIdx.y;
  if (idx < K * N) {
    int k = idx / N, n = idx - k * N;
    float v = src[(size_t)head * K * N + idx];
    dst[(size_t)head * K * N + ((size_t)(k >> 4) * N + n) * 16 + (k & 15)] =
        (__bf16)v;
  }
}

// ---------------------------------------------------------------------------
// Kernel 1: fused QKV projection.  grid = (BS/64, H, 3), block = 128 (4 waves)
//   out = x @ W[h] + bias[h];  Q,K stored [b,h,s,e]; V stored transposed [b,h,e,s]
//   W tiles (shared by all 4 waves) streamed via double-buffered async->LDS.
// ---------------------------------------------------------------------------
__global__ __launch_bounds__(128) void qkv_kernel(
    const __bf16* __restrict__ xb,
    const __bf16* __restrict__ WQt, const __bf16* __restrict__ WKt,
    const __bf16* __restrict__ WVt,
    const float* __restrict__ bQ, const float* __restrict__ bK,
    const float* __restrict__ bV,
    __bf16* __restrict__ Qb, __bf16* __restrict__ Kb, __bf16* __restrict__ Vt) {
  // sw[buf][k-group][n][i] : one 32-key x 64-col W tile per buffer
  __shared__ __bf16 sw[2][2][Eq][16];
  int t = threadIdx.x;
  int lane = t & 31, wave = t >> 5;
  int row0 = blockIdx.x * 64 + wave * 16;   // global row in [0, B*S)
  int h = blockIdx.y, which = blockIdx.z;

  const __bf16* Wt = (which == 0 ? WQt : which == 1 ? WKt : WVt) +
                     (size_t)h * Dq * Eq;   // fragment layout [D/16][64][16]
  const float* bias = (which == 0 ? bQ : which == 1 ? bK : bV) + h * Eq;

  auto issue = [&](int buf, int kb) {
    int g = kb >> 4;                         // copies k-groups g and g+1
#pragma unroll
    for (int grp = 0; grp < 2; ++grp) {
      const __bf16* src = Wt + (size_t)(g + grp) * (Eq * 16) + t * 8;
      uint32_t lds = (uint32_t)(uintptr_t)(&sw[buf][grp][0][0]) + t * 16;
      async_copy16(lds, src);
    }
  };

  issue(0, 0);
  v8f acc[4] = {};
  int n = lane & 15, kh = lane >> 4;
  for (int kb = 0; kb < Dq; kb += 32) {
    int buf = (kb >> 5) & 1;
    wait_async0();
    __syncthreads();
    if (kb + 32 < Dq) issue(buf ^ 1, kb + 32);
    v16bf a = load_a(xb, Dq, row0, kb, lane);
    // distinct registers for all four B fragments -> ds_loads can be clause'd
    // ahead of the WMMA chain instead of one full s_wait_dscnt per WMMA
    v16bf b0 = *(const v16bf*)(&sw[buf][kh][0 * 16 + n][0]);
    v16bf b1 = *(const v16bf*)(&sw[buf][kh][1 * 16 + n][0]);
    v16bf b2 = *(const v16bf*)(&sw[buf][kh][2 * 16 + n][0]);
    v16bf b3 = *(const v16bf*)(&sw[buf][kh][3 * 16 + n][0]);
    acc[0] = wmma_bf16(a, b0, acc[0]);
    acc[1] = wmma_bf16(a, b1, acc[1]);
    acc[2] = wmma_bf16(a, b2, acc[2]);
    acc[3] = wmma_bf16(a, b3, acc[3]);
  }

  int bb = row0 / Sq, s = row0 % Sq;
  if (which == 2) {
    // V^T store: [b,h,e,s], contiguous in s -> one 16B store per (et)
#pragma unroll
    for (int et = 0; et < 4; ++et) {
      int e = et * 16 + n;
      float bv = bias[e];
      v8bf vs;
#pragma unroll
      for (int i = 0; i < 8; ++i) vs[i] = (__bf16)(acc[et][i] + bv);
      *(v8bf*)(Vt + ((size_t)(bb * Hq + h) * Eq + e) * Sq + s + kh * 8) = vs;
    }
  } else {
    __bf16* out = (which == 0 ? Qb : Kb);
#pragma unroll
    for (int et = 0; et < 4; ++et) {
      float bv = bias[et * 16 + n];
#pragma unroll
      for (int i = 0; i < 8; ++i) {
        int srow = s + kh * 8 + i;
        out[((size_t)(bb * Hq + h) * Sq + srow) * Eq + et * 16 + n] =
            (__bf16)(acc[et][i] + bv);
      }
    }
  }
}

// ---------------------------------------------------------------------------
// Kernel 2: register-resident causal flash attention.
//   grid = (S/64, H, B), block = 128; each wave owns a 16-query tile and walks
//   its causal key range in 32-key steps (8 WMMAs/step).  Computes S^T = K.Q^T
//   so softmax reductions are in-lane + one shfl_xor(16); P re-fragmented to a
//   B operand with shfl_xor(16) only.  V is read from V^T (contiguous).
// ---------------------------------------------------------------------------
__global__ __launch_bounds__(128) void attn_kernel(
    const __bf16* __restrict__ Qb, const __bf16* __restrict__ Kb,
    const __bf16* __restrict__ Vt, __bf16* __restrict__ Zb) {
  int lane = threadIdx.x & 31, wave = threadIdx.x >> 5;
  int q0 = blockIdx.x * 64 + wave * 16;
  int h = blockIdx.y, bb = blockIdx.z;

  const __bf16* Qp = Qb + (size_t)(bb * Hq + h) * Sq * Eq;  // [s][e]
  const __bf16* Kp = Kb + (size_t)(bb * Hq + h) * Sq * Eq;  // [s][e]
  const __bf16* Vp = Vt + (size_t)(bb * Hq + h) * Eq * Sq;  // [e][s]

  v16bf bq0 = load_bT(Qp, Eq, 0, q0, lane);   // Q^T fragments, loop-invariant
  v16bf bq1 = load_bT(Qp, Eq, 32, q0, lane);

  v8f o[4] = {};                 // O^T accumulators: 4 e-tiles x (16e x 16q)
  float runM = -3.0e38f, runL = 0.0f;
  int q = q0 + (lane & 15);
  int kh = lane >> 4;
  v8f zeroc = {};

  int nkt = (q0 + 16 + 31) / 32; // causal: keys <= q0+15
  for (int kt = 0; kt < nkt; ++kt) {
    int kbase = kt * 32;
    // S^T tiles: rows = keys, cols = queries
    v16bf ak0a = load_a(Kp, Eq, kbase, 0, lane);
    v16bf ak0b = load_a(Kp, Eq, kbase, 32, lane);
    v16bf ak1a = load_a(Kp, Eq, kbase + 16, 0, lane);
    v16bf ak1b = load_a(Kp, Eq, kbase + 16, 32, lane);
    v8f s0 = wmma_bf16(ak0a, bq0, zeroc);
    s0 = wmma_bf16(ak0b, bq1, s0);
    v8f s1 = wmma_bf16(ak1a, bq0, zeroc);
    s1 = wmma_bf16(ak1b, bq1, s1);

    // scale + causal mask + online softmax (per-query = per-lane column)
    float p0[8], p1[8];
    float tmax = -3.0e38f;
#pragma unroll
    for (int i = 0; i < 8; ++i) {
      int key0 = kbase + kh * 8 + i;
      int key1 = key0 + 16;
      float v0 = s0[i] * 0.125f; if (key0 > q) v0 = IGNORE_VAL;
      float v1 = s1[i] * 0.125f; if (key1 > q) v1 = IGNORE_VAL;
      p0[i] = v0; p1[i] = v1;
      tmax = fmaxf(tmax, fmaxf(v0, v1));
    }
    tmax = fmaxf(tmax, __shfl_xor(tmax, 16));
    float newM = fmaxf(runM, tmax);
    float alpha = __expf(runM - newM);
    float lsum = 0.0f;
#pragma unroll
    for (int i = 0; i < 8; ++i) {
      p0[i] = __expf(p0[i] - newM);
      p1[i] = __expf(p1[i] - newM);
      lsum += p0[i] + p1[i];
    }
    lsum += __shfl_xor(lsum, 16);
    runL = runL * alpha + lsum;
    runM = newM;

    // Build B-fragment of P^T (32keys x 16q): lane<16 needs keys 0..15,
    // lane>=16 needs keys 16..31; missing halves come from the xor-16 partner.
    v16bf pb;
#pragma unroll
    for (int i = 0; i < 8; ++i) {
      float t0 = __shfl_xor(p0[i], 16);
      float t1 = __shfl_xor(p1[i], 16);
      pb[i]     = (__bf16)(lane < 16 ? p0[i] : t1);
      pb[8 + i] = (__bf16)(lane < 16 ? t0 : p1[i]);
    }

    // O^T(e,q) = alpha*O^T + V^T(e,k) . P^T(k,q)   (V^T is row-major [e][s])
    // hoisted V fragments -> global loads clause ahead of the WMMA chain
    v16bf av0 = load_a(Vp, Sq, 0, kbase, lane);
    v16bf av1 = load_a(Vp, Sq, 16, kbase, lane);
    v16bf av2 = load_a(Vp, Sq, 32, kbase, lane);
    v16bf av3 = load_a(Vp, Sq, 48, kbase, lane);
    o[0] = wmma_bf16(av0, pb, o[0] * alpha);
    o[1] = wmma_bf16(av1, pb, o[1] * alpha);
    o[2] = wmma_bf16(av2, pb, o[2] * alpha);
    o[3] = wmma_bf16(av3, pb, o[3] * alpha);
  }

  // Z stored [b, s, h*64+e] so out-proj reads contiguous K; 16B vector stores
  float inv = 1.0f / runL;
#pragma unroll
  for (int et = 0; et < 4; ++et) {
    v8bf zs;
#pragma unroll
    for (int i = 0; i < 8; ++i) zs[i] = (__bf16)(o[et][i] * inv);
    *(v8bf*)(Zb + (((size_t)bb * Sq + q) * Hq + h) * Eq + et * 16 + kh * 8) = zs;
  }
}

// ---------------------------------------------------------------------------
// Kernel 3: output projection.  grid = (BS/64, D/64), block = 128.
//   out[bs,d] = sum_{he} Z[bs,he] * W_O[he,d] + b_O[d]   (f32 output)
//   W_O tiles streamed via double-buffered async->LDS (shared by 4 waves).
// ---------------------------------------------------------------------------
__global__ __launch_bounds__(128) void oproj_kernel(
    const __bf16* __restrict__ Zb, const __bf16* __restrict__ WOt,
    const float* __restrict__ bO, float* __restrict__ out) {
  __shared__ __bf16 sw[2][2][Eq][16];
  int t = threadIdx.x;
  int lane = t & 31, wave = t >> 5;
  int row0 = blockIdx.x * 64 + wave * 16;
  int n0 = blockIdx.y * 64;

  auto issue = [&](int buf, int kb) {
    int g = kb >> 4;
#pragma unroll
    for (int grp = 0; grp < 2; ++grp) {
      const __bf16* src = WOt + ((size_t)(g + grp) * Dq + n0) * 16 + t * 8;
      uint32_t lds = (uint32_t)(uintptr_t)(&sw[buf][grp][0][0]) + t * 16;
      async_copy16(lds, src);
    }
  };

  issue(0, 0);
  v8f acc[4] = {};
  int n = lane & 15, kh = lane >> 4;
  for (int kb = 0; kb < Dq; kb += 32) {
    int buf = (kb >> 5) & 1;
    wait_async0();
    __syncthreads();
    if (kb + 32 < Dq) issue(buf ^ 1, kb + 32);
    v16bf a = load_a(Zb, Dq, row0, kb, lane);
    v16bf b0 = *(const v16bf*)(&sw[buf][kh][0 * 16 + n][0]);
    v16bf b1 = *(const v16bf*)(&sw[buf][kh][1 * 16 + n][0]);
    v16bf b2 = *(const v16bf*)(&sw[buf][kh][2 * 16 + n][0]);
    v16bf b3 = *(const v16bf*)(&sw[buf][kh][3 * 16 + n][0]);
    acc[0] = wmma_bf16(a, b0, acc[0]);
    acc[1] = wmma_bf16(a, b1, acc[1]);
    acc[2] = wmma_bf16(a, b2, acc[2]);
    acc[3] = wmma_bf16(a, b3, acc[3]);
  }

#pragma unroll
  for (int nt = 0; nt < 4; ++nt) {
    int d = n0 + nt * 16 + n;
    float bv = bO[d];
#pragma unroll
    for (int i = 0; i < 8; ++i)
      out[(size_t)(row0 + kh * 8 + i) * Dq + d] = acc[nt][i] + bv;
  }
}

// ---------------------------------------------------------------------------
extern "C" void kernel_launch(void* const* d_in, const int* in_sizes, int n_in,
                              void* d_out, int out_size, void* d_ws, size_t ws_size,
                              hipStream_t stream) {
  (void)in_sizes; (void)n_in; (void)out_size; (void)ws_size;
  const float* x   = (const float*)d_in[0];
  const float* W_Q = (const float*)d_in[1];
  const float* b_Q = (const float*)d_in[2];
  const float* W_K = (const float*)d_in[3];
  const float* b_K = (const float*)d_in[4];
  const float* W_V = (const float*)d_in[5];
  const float* b_V = (const float*)d_in[6];
  const float* W_O = (const float*)d_in[7];
  const float* b_O = (const float*)d_in[8];
  float* out = (float*)d_out;

  // Workspace carve-up (bf16): xb 8MB, 4 weights 8MB, Q/K/Vt 24MB, Z 8MB = 48MB
  char* w = (char*)d_ws;
  auto carve = [&](size_t elems) { __bf16* p = (__bf16*)w; w += elems * sizeof(__bf16); return p; };
  __bf16* xb  = carve((size_t)BSq * Dq);
  __bf16* WQt = carve((size_t)Hq * Dq * Eq);
  __bf16* WKt = carve((size_t)Hq * Dq * Eq);
  __bf16* WVt = carve((size_t)Hq * Dq * Eq);
  __bf16* WOt = carve((size_t)Hq * Eq * Dq);
  __bf16* Qb  = carve((size_t)Bq * Hq * Sq * Eq);
  __bf16* Kb  = carve((size_t)Bq * Hq * Sq * Eq);
  __bf16* Vt  = carve((size_t)Bq * Hq * Eq * Sq);
  __bf16* Zb  = carve((size_t)Bq * Sq * Hq * Eq);

  { // x: linear bf16 convert
    int n4 = (BSq * Dq) / 4;
    cvt_bf16_kernel<<<(n4 + 255) / 256, 256, 0, stream>>>((const float4*)x, xb, n4);
  }
  // weights: fragment-native layout [head][K/16][N][16]
  cvt_wfrag_kernel<<<dim3((Dq * Eq + 255) / 256, Hq), 256, 0, stream>>>(W_Q, WQt, Dq, Eq);
  cvt_wfrag_kernel<<<dim3((Dq * Eq + 255) / 256, Hq), 256, 0, stream>>>(W_K, WKt, Dq, Eq);
  cvt_wfrag_kernel<<<dim3((Dq * Eq + 255) / 256, Hq), 256, 0, stream>>>(W_V, WVt, Dq, Eq);
  cvt_wfrag_kernel<<<dim3((Dq * Dq + 255) / 256, 1), 256, 0, stream>>>(W_O, WOt, Dq, Dq);

  qkv_kernel<<<dim3(BSq / 64, Hq, 3), 128, 0, stream>>>(
      xb, WQt, WKt, WVt, b_Q, b_K, b_V, Qb, Kb, Vt);

  attn_kernel<<<dim3(Sq / 64, Hq, Bq), 128, 0, stream>>>(Qb, Kb, Vt, Zb);

  oproj_kernel<<<dim3(BSq / 64, Dq / 64), 128, 0, stream>>>(Zb, WOt, b_O, out);
}
